// GraphModule_24309514896019
// MI455X (gfx1250) — compile-verified, matrix operand order checked by reading, not run on hardware
//
#include <hip/hip_runtime.h>
#include <hip/hip_bf16.h>

// ---------------------------------------------------------------------------
// 2-layer GAT for MI455X (gfx1250, wave32).
//  - GEMMs use v_wmma_f32_16x16x32_f16; W staged in LDS pre-swizzled into
//    per-lane fragment order so B loads are 2x ds_load_b128.
//  - A fragments hoisted out of the column-tile loop (loaded once per wave).
//  - Full-tile fast path: branch-free store epilogue w/ immediate offsets.
//  - Edge softmax: atomic-max (ordered-uint encode) -> atomic exp-sum ->
//    alpha-weighted scatter-add (global_atomic_add_f32).
// ---------------------------------------------------------------------------

typedef __attribute__((ext_vector_type(16))) _Float16 v16h;
typedef __attribute__((ext_vector_type(8)))  float    v8f;

#define NEG_SLOPE 0.2f

// Monotonic float<->uint mapping so atomicMax(u32) implements float max.
__device__ __forceinline__ unsigned fenc(float f) {
    unsigned u = __float_as_uint(f);
    return (u & 0x80000000u) ? ~u : (u | 0x80000000u);
}
__device__ __forceinline__ float fdec(unsigned u) {
    return (u & 0x80000000u) ? __uint_as_float(u & 0x7fffffffu)
                             : __uint_as_float(~u);
}
#define NEG_INF_ENC 0x007FFFFFu   // fenc(-inf)

// ---------------------------------------------------------------------------
// fill kernel (zeros / -inf codes)
// ---------------------------------------------------------------------------
__global__ void fill_u32(unsigned* __restrict__ p, unsigned v, long long count) {
    long long i = (long long)blockIdx.x * blockDim.x + threadIdx.x;
    if (i < count) p[i] = v;
}

// ---------------------------------------------------------------------------
// GEMM: Out[nrows x NCOLS] = A[nrows x 128] @ W[128 x NCOLS]   (f32 in/out)
// 256-thread block = 8 waves; wave owns a 16-row tile, loops over NCOLS/16
// column tiles; K=128 consumed as 4 x v_wmma_f32_16x16x32_f16.
//
// LDS holds W as f16 in FRAGMENT ORDER: Wfrag[(ct*4 + kt)*32 + lane] is the
// 16 contiguous halves lane `lane` feeds to the WMMA for column-tile ct,
// k-tile kt.  B-matrix lane layout (mirror of 16-bit A layout):
//   lane = (half<<4) | (n & 15), element e -> K = kt*32 + (e<8?0:16) + half*8 + (e&7)
// ---------------------------------------------------------------------------
template <int NCOLS>
__global__ __launch_bounds__(256) void gemm_wmma(const float* __restrict__ A,
                                                 const float* __restrict__ W,
                                                 float* __restrict__ Out,
                                                 int nrows) {
    constexpr int CT = NCOLS / 16;             // column tiles
    __shared__ v16h Wfrag[CT * 4 * 32];        // 32B per lane-fragment
    {
        _Float16* wl = (_Float16*)Wfrag;
        for (int i = threadIdx.x; i < 128 * NCOLS; i += 256) {
            const int k  = i / NCOLS, nn = i % NCOLS;
            const int kt = k >> 5, kk = k & 31;
            const int half = (kk >> 3) & 1;
            const int elem = ((kk >> 4) << 3) | (kk & 7);
            const int lane = (half << 4) | (nn & 15);
            const int ct   = nn >> 4;
            wl[((((ct << 2) + kt) << 5) + lane) * 16 + elem] = (_Float16)W[i];
        }
    }
    __syncthreads();

    const int wave = threadIdx.x >> 5;
    const int lane = threadIdx.x & 31;
    const int row0 = (blockIdx.x * 8 + wave) * 16;
    if (row0 >= nrows) return;

    const int  half = lane >> 4;    // 0: K {0..7,16..23}; 1: K {8..15,24..31}
    const int  l15  = lane & 15;
    const int  m    = row0 + l15;   // A row for this lane (both half-groups)
    const bool fulltile = (row0 + 16 <= nrows);   // wave-uniform
    const bool mvalid   = fulltile || (m < nrows);
    const float* arow = A + (size_t)(mvalid ? m : 0) * 128;

    // A fragments depend only on kt -> load once, reuse across column tiles.
    v16h afrag[4];
#pragma unroll
    for (int kt = 0; kt < 4; ++kt) {
        v16h a = {};
        if (mvalid) {
            const int K0 = kt * 32 + half * 8;
#pragma unroll
            for (int j = 0; j < 8; ++j) {
                a[j]     = (_Float16)arow[K0 + j];
                a[8 + j] = (_Float16)arow[K0 + 16 + j];
            }
        }
        afrag[kt] = a;
    }

#pragma unroll
    for (int ct = 0; ct < CT; ++ct) {
        v8f acc = {};
#pragma unroll
        for (int kt = 0; kt < 4; ++kt) {
            const v16h b = Wfrag[(((ct << 2) + kt) << 5) + lane];  // 2x ds_load_b128
            acc = __builtin_amdgcn_wmma_f32_16x16x32_f16(
                false, afrag[kt], false, b, (short)0, acc, false, false);
        }
        // C/D layout: VGPR v -> M = row0 + v + 8*half, N = ct*16 + (lane&15)
        float* orow = Out + (size_t)(row0 + half * 8) * NCOLS + (ct * 16 + l15);
        if (fulltile) {
#pragma unroll
            for (int v = 0; v < 8; ++v)
                orow[v * NCOLS] = acc[v];      // imm-offset store clause
        } else {
#pragma unroll
            for (int v = 0; v < 8; ++v)
                if (row0 + half * 8 + v < nrows) orow[v * NCOLS] = acc[v];
        }
    }
}

// ---------------------------------------------------------------------------
// Per-node attention scores: s_src[n,h] = <feat[n,h,:], a_src[h,:]>  (same dst)
// ---------------------------------------------------------------------------
__global__ void scores_kernel(const float* __restrict__ feat,
                              const float* __restrict__ a_src,
                              const float* __restrict__ a_dst,
                              float* __restrict__ ssrc,
                              float* __restrict__ sdst,
                              int n, int H, int C) {
    int i = blockIdx.x * blockDim.x + threadIdx.x;
    if (i >= n) return;
    const float* f = feat + (size_t)i * H * C;
    for (int h = 0; h < H; ++h) {
        float ss = 0.f, sd = 0.f;
        for (int c = 0; c < C; ++c) {
            float v = f[h * C + c];
            ss += v * a_src[h * C + c];
            sd += v * a_dst[h * C + c];
        }
        ssrc[i * H + h] = ss;
        sdst[i * H + h] = sd;
    }
}

__device__ __forceinline__ void edge_nodes(const int* __restrict__ ei,
                                           int E, int e, int& s, int& d) {
    if (e < E) { s = ei[e]; d = ei[E + e]; }
    else       { s = e - E; d = e - E;     }   // implicit self-loop
}

// Pass 1: segment max of leaky-relu edge logits (atomicMax on ordered uint)
__global__ void edge_max(const int* __restrict__ ei, int E, int ET,
                         const float* __restrict__ ssrc,
                         const float* __restrict__ sdst,
                         unsigned* __restrict__ mEnc, int H) {
    int e = blockIdx.x * blockDim.x + threadIdx.x;
    if (e >= ET) return;
    int s, d; edge_nodes(ei, E, e, s, d);
    for (int h = 0; h < H; ++h) {
        float ev = ssrc[s * H + h] + sdst[d * H + h];
        ev = ev > 0.f ? ev : NEG_SLOPE * ev;
        atomicMax(&mEnc[d * H + h], fenc(ev));
    }
}

// Pass 2: z[d,h] += exp(e - m[d,h])
__global__ void edge_sum(const int* __restrict__ ei, int E, int ET,
                         const float* __restrict__ ssrc,
                         const float* __restrict__ sdst,
                         const unsigned* __restrict__ mEnc,
                         float* __restrict__ z, int H) {
    int e = blockIdx.x * blockDim.x + threadIdx.x;
    if (e >= ET) return;
    int s, d; edge_nodes(ei, E, e, s, d);
    for (int h = 0; h < H; ++h) {
        float ev = ssrc[s * H + h] + sdst[d * H + h];
        ev = ev > 0.f ? ev : NEG_SLOPE * ev;
        float m = fdec(mEnc[d * H + h]);
        atomicAdd(&z[d * H + h], expf(ev - m));
    }
}

// Pass 3: agg[d,h,:] += alpha * feat[s,h,:]
__global__ void edge_agg(const int* __restrict__ ei, int E, int ET,
                         const float* __restrict__ ssrc,
                         const float* __restrict__ sdst,
                         const unsigned* __restrict__ mEnc,
                         const float* __restrict__ z,
                         const float* __restrict__ feat,
                         float* __restrict__ agg, int H, int C) {
    int e = blockIdx.x * blockDim.x + threadIdx.x;
    if (e >= ET) return;
    int s, d; edge_nodes(ei, E, e, s, d);
    float alpha[4];
    for (int h = 0; h < H; ++h) {
        float ev = ssrc[s * H + h] + sdst[d * H + h];
        ev = ev > 0.f ? ev : NEG_SLOPE * ev;
        float m = fdec(mEnc[d * H + h]);
        alpha[h] = expf(ev - m) / z[d * H + h];
    }
    const int HC = H * C;
    const float* fs = feat + (size_t)s * HC;
    float* ad = agg + (size_t)d * HC;
    for (int h = 0; h < H; ++h) {
        float a = alpha[h];
        for (int c = 0; c < C; ++c)
            atomicAdd(&ad[h * C + c], fs[h * C + c] * a);
    }
}

// ELU(x + bias) in place over [n, d]
__global__ void elu_bias(float* __restrict__ x, const float* __restrict__ b,
                         long long total, int d) {
    long long i = (long long)blockIdx.x * blockDim.x + threadIdx.x;
    if (i >= total) return;
    float v = x[i] + b[(int)(i % d)];
    x[i] = v > 0.f ? v : (expf(v) - 1.f);
}

// out[n,:] = (out[n,:] + b2) / max(||out[n,:] + b2||, 1e-12)   in place
__global__ void norm_out(float* __restrict__ out, const float* __restrict__ b2,
                         int n, int d) {
    int i = blockIdx.x * blockDim.x + threadIdx.x;
    if (i >= n) return;
    float* o = out + (size_t)i * d;
    float ss = 0.f;
    for (int c = 0; c < d; ++c) { float v = o[c] + b2[c]; ss += v * v; }
    float nm  = fmaxf(sqrtf(ss), 1e-12f);
    float inv = 1.f / nm;
    for (int c = 0; c < d; ++c) o[c] = (o[c] + b2[c]) * inv;
}

// ---------------------------------------------------------------------------
// host launcher
// ---------------------------------------------------------------------------
extern "C" void kernel_launch(void* const* d_in, const int* in_sizes, int n_in,
                              void* d_out, int out_size, void* d_ws, size_t ws_size,
                              hipStream_t stream) {
    (void)n_in; (void)out_size; (void)ws_size;
    const float* x   = (const float*)d_in[0];
    const int*   ei  = (const int*)d_in[1];
    const float* W1  = (const float*)d_in[2];
    const float* as1 = (const float*)d_in[3];
    const float* ad1 = (const float*)d_in[4];
    const float* b1  = (const float*)d_in[5];
    const float* W2  = (const float*)d_in[6];
    const float* as2 = (const float*)d_in[7];
    const float* ad2 = (const float*)d_in[8];
    const float* b2  = (const float*)d_in[9];
    float* out = (float*)d_out;

    const int IN = 128, H1 = 4, C1 = 32, HC1 = 128, C2 = 64;
    const int n  = in_sizes[0] / IN;   // 50000
    const int e  = in_sizes[1] / 2;    // 800000
    const int et = e + n;              // + self loops

    // bump-allocate workspace (~67 MB)
    char* p = (char*)d_ws;
    auto alloc = [&](size_t bytes) -> void* {
        void* r = p; p += (bytes + 255) & ~(size_t)255; return r;
    };
    float*    h1   = (float*)   alloc((size_t)n * HC1 * 4);
    float*    agg1 = (float*)   alloc((size_t)n * HC1 * 4);
    float*    h2   = (float*)   alloc((size_t)n * C2  * 4);
    float*    ss1  = (float*)   alloc((size_t)n * H1  * 4);
    float*    sd1  = (float*)   alloc((size_t)n * H1  * 4);
    unsigned* m1   = (unsigned*)alloc((size_t)n * H1  * 4);
    float*    z1   = (float*)   alloc((size_t)n * H1  * 4);
    float*    ss2  = (float*)   alloc((size_t)n * 4);
    float*    sd2  = (float*)   alloc((size_t)n * 4);
    unsigned* m2   = (unsigned*)alloc((size_t)n * 4);
    float*    z2   = (float*)   alloc((size_t)n * 4);

    auto nb = [](long long cnt, int bs) { return (unsigned)((cnt + bs - 1) / bs); };

    // init accumulators (replay-safe: everything re-initialized each call)
    fill_u32<<<nb((long long)n * HC1, 256), 256, 0, stream>>>((unsigned*)agg1, 0u, (long long)n * HC1);
    fill_u32<<<nb(n * H1, 256), 256, 0, stream>>>(m1, NEG_INF_ENC, n * H1);
    fill_u32<<<nb(n * H1, 256), 256, 0, stream>>>((unsigned*)z1, 0u, n * H1);
    fill_u32<<<nb(n, 256), 256, 0, stream>>>(m2, NEG_INF_ENC, n);
    fill_u32<<<nb(n, 256), 256, 0, stream>>>((unsigned*)z2, 0u, n);
    fill_u32<<<nb((long long)n * C2, 256), 256, 0, stream>>>((unsigned*)out, 0u, (long long)n * C2);

    // ---- layer 1: GATConv(128 -> 4 heads x 32), concat ----
    gemm_wmma<128><<<nb(n, 128), 256, 0, stream>>>(x, W1, h1, n);
    scores_kernel<<<nb(n, 256), 256, 0, stream>>>(h1, as1, ad1, ss1, sd1, n, H1, C1);
    edge_max<<<nb(et, 256), 256, 0, stream>>>(ei, e, et, ss1, sd1, m1, H1);
    edge_sum<<<nb(et, 256), 256, 0, stream>>>(ei, e, et, ss1, sd1, m1, z1, H1);
    edge_agg<<<nb(et, 256), 256, 0, stream>>>(ei, e, et, ss1, sd1, m1, z1, h1, agg1, H1, C1);
    elu_bias<<<nb((long long)n * HC1, 256), 256, 0, stream>>>(agg1, b1, (long long)n * HC1, HC1);

    // ---- layer 2: GATConv(128 -> 1 head x 64) ----
    gemm_wmma<64><<<nb(n, 128), 256, 0, stream>>>(agg1, W2, h2, n);
    scores_kernel<<<nb(n, 256), 256, 0, stream>>>(h2, as2, ad2, ss2, sd2, n, 1, C2);
    edge_max<<<nb(et, 256), 256, 0, stream>>>(ei, e, et, ss2, sd2, m2, 1);
    edge_sum<<<nb(et, 256), 256, 0, stream>>>(ei, e, et, ss2, sd2, m2, z2, 1);
    edge_agg<<<nb(et, 256), 256, 0, stream>>>(ei, e, et, ss2, sd2, m2, z2, h2, out, 1, C2);

    // mean over 1 head = identity; add bias + L2 normalize
    norm_out<<<nb(n, 256), 256, 0, stream>>>(out, b2, n, C2);
}
